// Decoder_53111565582609
// MI455X (gfx1250) — compile-verified, matrix operand order checked by reading, not run on hardware
//
#include <hip/hip_runtime.h>
#include <hip/hip_bf16.h>
#include <math.h>

typedef _Float16 v16h __attribute__((ext_vector_type(16)));
typedef _Float16 v8h  __attribute__((ext_vector_type(8)));
typedef float    v8f  __attribute__((ext_vector_type(8)));

#define B_TOT 256
#define NNODE 300
#define HDIM  256
#define NJ    20
#define NM    15
#define NHEAD 8
#define DK    32
#define BT    16      // batch rows per workgroup (WMMA M)
#define KPAD  288     // fuse_W1 K (262) padded to multiple of 32
#define NEGV  (-1e9f)
#define CLIPV 10.0f

// ---------------- WMMA helpers (16x16x32 f16 -> f32) ----------------
// A fragment (16x32): lane&15 = row, (lane>>4)*8 = K base; halves [kb..kb+7]
// in v[0..3] and [kb+16..kb+23] in v[4..7]. B fragment for X @ W^T is the
// mirrored layout -> same loader applied to row-major W (row = out col).
static __device__ __forceinline__ v16h load_frag(const _Float16* base, int stride, int lane) {
  int r  = lane & 15;
  int kb = (lane >> 4) << 3;
  const _Float16* p = base + (size_t)r * stride + kb;
  v8h lo = *(const v8h*)p;
  v8h hi = *(const v8h*)(p + 16);
  v16h a;
#pragma unroll
  for (int i = 0; i < 8; ++i) { a[i] = lo[i]; a[i + 8] = hi[i]; }
  return a;
}

static __device__ __forceinline__ v8f wmma16(v16h a, v16h b, v8f c) {
  return __builtin_amdgcn_wmma_f32_16x16x32_f16(false, a, false, b, (short)0, c, false, false);
}

static __device__ __forceinline__ void store_tile_f32(float* base, int stride, int lane, v8f d) {
  int c = lane & 15, r0 = (lane >> 4) << 3;
#pragma unroll
  for (int i = 0; i < 8; ++i) base[(size_t)(r0 + i) * stride + c] = d[i];
}

static __device__ __forceinline__ void store_tile_f16(_Float16* base, int stride, int lane, v8f d) {
  int c = lane & 15, r0 = (lane >> 4) << 3;
#pragma unroll
  for (int i = 0; i < 8; ++i) base[(size_t)(r0 + i) * stride + c] = (_Float16)d[i];
}

// ---------------- weight conversion kernels ----------------
__global__ void cvt_f32_f16(const float* __restrict__ src, _Float16* __restrict__ dst, int n) {
  int i = blockIdx.x * blockDim.x + threadIdx.x;
  if (i < n) dst[i] = (_Float16)src[i];
}

__global__ void cvt_pad_f16(const float* __restrict__ src, _Float16* __restrict__ dst,
                            int rows, int k, int kpad) {
  int i = blockIdx.x * blockDim.x + threadIdx.x;
  int tot = rows * kpad;
  if (i < tot) {
    int r = i / kpad, c = i - r * kpad;
    dst[i] = (c < k) ? (_Float16)src[r * k + c] : (_Float16)0.f;
  }
}

// ---------------- precompute: kl = E@Wpk^T, ks = E@Wk^T, vs = E@Wv^T ----------------
__global__ void precompute_kernel(const float* __restrict__ enc,
                                  const _Float16* __restrict__ wpk,
                                  const _Float16* __restrict__ wk,
                                  const _Float16* __restrict__ wv,
                                  _Float16* __restrict__ kl,
                                  _Float16* __restrict__ ks,
                                  _Float16* __restrict__ vs) {
  __shared__ _Float16 x16[BT][HDIM];
  const int tid = threadIdx.x, wave = tid >> 5, lane = tid & 31;
  const size_t row0 = (size_t)blockIdx.x * BT;
  for (int idx = tid; idx < BT * HDIM; idx += 256) {
    int r = idx >> 8, c = idx & 255;
    x16[r][c] = (_Float16)enc[(row0 + r) * HDIM + c];
  }
  __syncthreads();
  for (int jb = wave; jb < 48; jb += 8) {        // 3 matrices x 16 col-tiles
    int mat = jb >> 4, ct = jb & 15;
    const _Float16* W = (mat == 0) ? wpk : ((mat == 1) ? wk : wv);
    _Float16* out     = (mat == 0) ? kl  : ((mat == 1) ? ks : vs);
    v8f d = {};
    for (int kk = 0; kk < HDIM; kk += 32) {
      v16h a = load_frag(&x16[0][kk], HDIM, lane);
      v16h b = load_frag(W + (size_t)(ct * 16) * HDIM + kk, HDIM, lane);
      d = wmma16(a, b, d);
    }
    store_tile_f16(out + row0 * HDIM + ct * 16, HDIM, lane, d);
  }
}

// ---------------- persistent decoder ----------------
struct SmemDec {
  float     hx[BT][HDIM];
  _Float16  hx16[BT][HDIM];
  _Float16  x16[BT][HDIM];
  _Float16  q16[BT][HDIM];
  _Float16  att16[BT][HDIM];
  float     glimpse[BT][HDIM];
  float     fused[BT][HDIM];
  _Float16  h1[BT][HDIM];
  _Float16  cin[BT][KPAD];
  float     ucand[BT][NJ];
  float     jready[BT][NJ];
  float     mavail[BT][NM];
  int       counts[BT][NJ];
  int       safeidx[BT][NJ];
  float     dyn[BT][NJ][6];
  float     logp_acc[BT];
  float     ent_acc[BT];
  int       selglob[BT];
  union Scr {
    struct G { float r[BT][HDIM]; float z[BT][HDIM]; float inn[BT][HDIM]; float hnn[BT][HDIM]; } g;
    struct A { float m[BT * NHEAD][2]; float s[BT * NHEAD][2]; float acc[BT * NHEAD][2][DK]; } a;
    float dotp[BT][16];
  } scr;
};

__global__ void decoder_kernel(
    const float* __restrict__ enc, const float* __restrict__ proc_t,
    const float* __restrict__ start_tok,
    const float* __restrict__ b_ih, const float* __restrict__ b_hh,
    const float* __restrict__ fb1, const float* __restrict__ fb2,
    const unsigned char* __restrict__ maskpad, const int* __restrict__ opmach,
    const _Float16* __restrict__ Wih, const _Float16* __restrict__ Whh,
    const _Float16* __restrict__ Wq, const _Float16* __restrict__ Wo,
    const _Float16* __restrict__ FW1, const _Float16* __restrict__ FW2,
    const _Float16* __restrict__ kl, const _Float16* __restrict__ ks,
    const _Float16* __restrict__ vs,
    float* __restrict__ out) {
  extern __shared__ char smem_raw[];
  SmemDec& s = *reinterpret_cast<SmemDec*>(smem_raw);
  const int tid = threadIdx.x, wave = tid >> 5, lane = tid & 31;
  const int b0 = blockIdx.x * BT;
  const float inv_sqrt_dk = 0.17677669529663687f;  // 1/sqrt(32)
  const float inv_sqrt_h  = 0.0625f;               // 1/sqrt(256)

  // ---- init: hx = masked mean of encoder_out; x = start_token; state = 0 ----
  for (int idx = tid; idx < BT * HDIM; idx += 256) {
    int b = idx >> 8, c = idx & 255;
    float sum = 0.f, cnt = 0.f;
    for (int n = 0; n < NNODE; ++n) {
      float valid = maskpad[(size_t)(b0 + b) * NNODE + n] ? 0.f : 1.f;
      sum += valid * enc[((size_t)(b0 + b) * NNODE + n) * HDIM + c];
      cnt += valid;
    }
    float h0 = sum / fmaxf(cnt, 1.f);
    s.hx[b][c] = h0; s.hx16[b][c] = (_Float16)h0;
    s.x16[b][c] = (_Float16)start_tok[c];
  }
  if (tid < BT) {
    for (int j = 0; j < NJ; ++j) { s.counts[tid][j] = 0; s.jready[tid][j] = 0.f; }
    for (int m = 0; m < NM; ++m) s.mavail[tid][m] = 0.f;
    s.logp_acc[tid] = 0.f; s.ent_acc[tid] = 0.f;
  }
  __syncthreads();

  for (int t = 0; t < NNODE; ++t) {
    // ============ GRU: gi = x@Wih^T + bih ; gh = hx@Whh^T + bhh ============
    for (int jb = wave; jb < 48; jb += 8) {      // 3 gates x 16 col-tiles
      int typ = jb >> 4, ct = jb & 15;
      int colg = typ * HDIM + ct * 16;
      v8f di = {}, dh = {};
      for (int kk = 0; kk < HDIM; kk += 32) {
        v16h ax = load_frag(&s.x16[0][kk], HDIM, lane);
        v16h bi = load_frag(Wih + (size_t)colg * HDIM + kk, HDIM, lane);
        di = wmma16(ax, bi, di);
        v16h ah = load_frag(&s.hx16[0][kk], HDIM, lane);
        v16h bh = load_frag(Whh + (size_t)colg * HDIM + kk, HDIM, lane);
        dh = wmma16(ah, bh, dh);
      }
      int c = lane & 15, r0 = (lane >> 4) << 3;
      float bias_i = b_ih[colg + c], bias_h = b_hh[colg + c];
      int cl = ct * 16 + c;
#pragma unroll
      for (int i = 0; i < 8; ++i) {
        float vi = di[i] + bias_i, vh = dh[i] + bias_h;
        int rr = r0 + i;
        if (typ == 0)      s.scr.g.r[rr][cl] = 1.f / (1.f + __expf(-(vi + vh)));
        else if (typ == 1) s.scr.g.z[rr][cl] = 1.f / (1.f + __expf(-(vi + vh)));
        else             { s.scr.g.inn[rr][cl] = vi; s.scr.g.hnn[rr][cl] = vh; }
      }
    }
    __syncthreads();
    for (int idx = tid; idx < BT * HDIM; idx += 256) {
      int b = idx >> 8, c = idx & 255;
      float r = s.scr.g.r[b][c], z = s.scr.g.z[b][c];
      float nn = tanhf(s.scr.g.inn[b][c] + r * s.scr.g.hnn[b][c]);
      float h = (1.f - z) * nn + z * s.hx[b][c];
      s.hx[b][c] = h; s.hx16[b][c] = (_Float16)h;
    }
    __syncthreads();

    // ============ q = hx @ Wq^T ============
    for (int ct = wave; ct < 16; ct += 8) {
      v8f d = {};
      for (int kk = 0; kk < HDIM; kk += 32) {
        v16h a = load_frag(&s.hx16[0][kk], HDIM, lane);
        v16h b = load_frag(Wq + (size_t)(ct * 16) * HDIM + kk, HDIM, lane);
        d = wmma16(a, b, d);
      }
      store_tile_f16(&s.q16[0][ct * 16], HDIM, lane, d);
    }
    __syncthreads();

    // ============ multi-head pointer attention (online softmax) ============
    {
      int unit = tid >> 1, half = tid & 1;
      int b = unit >> 3, h = unit & 7;
      float qv[DK];
#pragma unroll
      for (int d = 0; d < DK; ++d) qv[d] = (float)s.q16[b][h * DK + d] * inv_sqrt_dk;
      float m = -3.0e38f, ssum = 0.f, acc[DK];
#pragma unroll
      for (int d = 0; d < DK; ++d) acc[d] = 0.f;
      for (int n = half; n < NNODE; n += 2) {
        const _Float16* kp = ks + ((size_t)(b0 + b) * NNODE + n) * HDIM + h * DK;
        const _Float16* vp = vs + ((size_t)(b0 + b) * NNODE + n) * HDIM + h * DK;
        __builtin_prefetch(kp + 2 * HDIM, 0, 1);   // global_prefetch_b8
        float sc = 0.f;
#pragma unroll
        for (int d = 0; d < DK; ++d) sc += qv[d] * (float)kp[d];
        if (maskpad[(size_t)(b0 + b) * NNODE + n]) sc = NEGV;
        float mn = fmaxf(m, sc);
        float so = __expf(m - mn), p = __expf(sc - mn);
#pragma unroll
        for (int d = 0; d < DK; ++d) acc[d] = acc[d] * so + p * (float)vp[d];
        ssum = ssum * so + p;
        m = mn;
      }
      s.scr.a.m[unit][half] = m; s.scr.a.s[unit][half] = ssum;
#pragma unroll
      for (int d = 0; d < DK; ++d) s.scr.a.acc[unit][half][d] = acc[d];
    }
    __syncthreads();
    if ((tid & 1) == 0) {
      int unit = tid >> 1; int b = unit >> 3, h = unit & 7;
      float m0 = s.scr.a.m[unit][0], m1 = s.scr.a.m[unit][1];
      float mt = fmaxf(m0, m1);
      float e0 = __expf(m0 - mt), e1 = __expf(m1 - mt);
      float inv = 1.f / (s.scr.a.s[unit][0] * e0 + s.scr.a.s[unit][1] * e1);
#pragma unroll
      for (int d = 0; d < DK; ++d) {
        float v = (s.scr.a.acc[unit][0][d] * e0 + s.scr.a.acc[unit][1][d] * e1) * inv;
        s.att16[b][h * DK + d] = (_Float16)v;
      }
    }
    __syncthreads();

    // ============ glimpse = att @ Wo^T ============
    for (int ct = wave; ct < 16; ct += 8) {
      v8f d = {};
      for (int kk = 0; kk < HDIM; kk += 32) {
        v16h a = load_frag(&s.att16[0][kk], HDIM, lane);
        v16h b = load_frag(Wo + (size_t)(ct * 16) * HDIM + kk, HDIM, lane);
        d = wmma16(a, b, d);
      }
      store_tile_f32(&s.glimpse[0][ct * 16], HDIM, lane, d);
    }
    __syncthreads();

    // ============ candidate dynamic features ============
    if (tid < BT) {
      int b = tid;
      float nrm = 1.f;
      for (int m = 0; m < NM; ++m) nrm = fmaxf(nrm, s.mavail[b][m]);
      float pts[NJ], ess[NJ], mas[NJ], ptsum = 0.f;
      for (int j = 0; j < NJ; ++j) {
        int cg = j * NM + s.counts[b][j];
        int safe = (cg < NNODE - 1) ? cg : (NNODE - 1);
        s.safeidx[b][j] = safe;
        float pt  = proc_t[(size_t)(b0 + b) * NNODE + safe];
        int  mach = opmach[(size_t)(b0 + b) * NNODE + safe];
        float ma = s.mavail[b][mach], jr = s.jready[b][j];
        float es = fmaxf(jr, ma);
        pts[j] = pt; ess[j] = es; mas[j] = ma; ptsum += pt;
      }
      float ptmean = ptsum / (float)NJ + 1e-5f;
      for (int j = 0; j < NJ; ++j) {
        float jr = s.jready[b][j];
        s.dyn[b][j][0] = ess[j] / nrm;
        s.dyn[b][j][1] = (ess[j] + pts[j]) / nrm;
        s.dyn[b][j][2] = (ess[j] - jr) / nrm;
        s.dyn[b][j][3] = (ess[j] - mas[j]) / nrm;
        s.dyn[b][j][4] = (float)(NM - s.counts[b][j]) / (float)NM;
        s.dyn[b][j][5] = pts[j] / ptmean;
      }
    }
    __syncthreads();

    // ============ fused MLP + pointer logits, per job j ============
    for (int j = 0; j < NJ; ++j) {
      for (int idx = tid; idx < BT * KPAD; idx += 256) {
        int b = idx / KPAD, c = idx - b * KPAD;
        _Float16 v;
        if (c < HDIM)          v = kl[((size_t)(b0 + b) * NNODE + s.safeidx[b][j]) * HDIM + c];
        else if (c < HDIM + 6) v = (_Float16)s.dyn[b][j][c - HDIM];
        else                   v = (_Float16)0.f;
        s.cin[b][c] = v;
      }
      __syncthreads();
      for (int ct = wave; ct < 16; ct += 8) {     // GEMM1 [16,288]x[288,256] + relu
        v8f d = {};
        for (int kk = 0; kk < KPAD; kk += 32) {
          v16h a = load_frag(&s.cin[0][kk], KPAD, lane);
          v16h b = load_frag(FW1 + (size_t)(ct * 16) * KPAD + kk, KPAD, lane);
          d = wmma16(a, b, d);
        }
        int c = lane & 15, r0 = (lane >> 4) << 3;
        float bias = fb1[ct * 16 + c];
#pragma unroll
        for (int i = 0; i < 8; ++i) {
          float v = d[i] + bias;
          s.h1[r0 + i][ct * 16 + c] = (_Float16)(v > 0.f ? v : 0.f);
        }
      }
      __syncthreads();
      for (int ct = wave; ct < 16; ct += 8) {     // GEMM2 [16,256]x[256,256]
        v8f d = {};
        for (int kk = 0; kk < HDIM; kk += 32) {
          v16h a = load_frag(&s.h1[0][kk], HDIM, lane);
          v16h b = load_frag(FW2 + (size_t)(ct * 16) * HDIM + kk, HDIM, lane);
          d = wmma16(a, b, d);
        }
        int c = lane & 15, r0 = (lane >> 4) << 3;
        float bias = fb2[ct * 16 + c];
#pragma unroll
        for (int i = 0; i < 8; ++i) s.fused[r0 + i][ct * 16 + c] = d[i] + bias;
      }
      __syncthreads();
      {
        int b = tid >> 4, seg = tid & 15;
        float p = 0.f;
#pragma unroll
        for (int c = 0; c < 16; ++c) p += s.glimpse[b][seg * 16 + c] * s.fused[b][seg * 16 + c];
        s.scr.dotp[b][seg] = p;
      }
      __syncthreads();
      if (tid < BT) {
        float acc = 0.f;
        for (int seg = 0; seg < 16; ++seg) acc += s.scr.dotp[tid][seg];
        s.ucand[tid][j] = tanhf(acc * inv_sqrt_h) * CLIPV;
      }
      __syncthreads();
    }

    // ============ masking + categorical sampling + state update ============
    if (tid < BT) {
      int b = tid;
      bool allm = true;
      for (int j = 0; j < NJ; ++j) if (s.counts[b][j] < NM) { allm = false; break; }
      float uv[NJ], mx = -3.0e38f;
      for (int j = 0; j < NJ; ++j) {
        float v = s.ucand[b][j];
        if (!allm && s.counts[b][j] >= NM) v = NEGV;
        uv[j] = v; mx = fmaxf(mx, v);
      }
      float sum = 0.f;
      for (int j = 0; j < NJ; ++j) sum += __expf(uv[j] - mx);
      float lse = __logf(sum);
      // deterministic splitmix64 hash of (step, batch row)
      unsigned long long z = ((unsigned long long)(t + 1)) * 0x9E3779B97F4A7C15ull
                           ^ ((unsigned long long)(b0 + b) + 0x51ull) * 0xBF58476D1CE4E5B9ull;
      z ^= z >> 30; z *= 0xBF58476D1CE4E5B9ull;
      z ^= z >> 27; z *= 0x94D049BB133111EBull;
      z ^= z >> 31;
      float rnd = (float)(z >> 40) * (1.0f / 16777216.0f);
      float target = rnd * sum;
      int sel = NJ - 1; float cdf = 0.f;
      for (int j = 0; j < NJ; ++j) {
        cdf += __expf(uv[j] - mx);
        if (cdf >= target) { sel = j; break; }
      }
      float logp = uv[sel] - mx - lse;
      float ent = 0.f;
      for (int j = 0; j < NJ; ++j) {
        float lp = uv[j] - mx - lse;
        ent -= __expf(lp) * lp;
      }
      s.logp_acc[b] += logp;
      s.ent_acc[b]  += ent;
      int cnt_before = s.counts[b][sel];
      int safe = s.safeidx[b][sel];
      float pt  = proc_t[(size_t)(b0 + b) * NNODE + safe];
      int  mach = opmach[(size_t)(b0 + b) * NNODE + safe];
      float comp = fmaxf(s.jready[b][sel], s.mavail[b][mach]) + pt;
      s.jready[b][sel] = comp;
      s.mavail[b][mach] = comp;
      s.counts[b][sel] = cnt_before + 1;
      int sg = sel * NM + cnt_before;
      if (sg > NNODE - 1) sg = NNODE - 1;
      s.selglob[b] = sg;
      out[(size_t)(b0 + b) * NNODE + t] = (float)sel;
    }
    __syncthreads();
    // next GRU input = encoder_out[b, sel_glob]
    for (int idx = tid; idx < BT * HDIM; idx += 256) {
      int b = idx >> 8, c = idx & 255;
      s.x16[b][c] = (_Float16)enc[((size_t)(b0 + b) * NNODE + s.selglob[b]) * HDIM + c];
    }
    __syncthreads();
  }

  if (tid < BT) {
    out[(size_t)B_TOT * NNODE + b0 + tid]         = s.logp_acc[tid];
    out[(size_t)B_TOT * NNODE + B_TOT + b0 + tid] = s.ent_acc[tid];
  }
}

extern "C" void kernel_launch(void* const* d_in, const int* in_sizes, int n_in,
                              void* d_out, int out_size, void* d_ws, size_t ws_size,
                              hipStream_t stream) {
  const float* enc       = (const float*)d_in[0];
  const float* proc_t    = (const float*)d_in[1];
  const float* start_tok = (const float*)d_in[2];
  const float* Wih       = (const float*)d_in[3];
  const float* Whh       = (const float*)d_in[4];
  const float* bih       = (const float*)d_in[5];
  const float* bhh       = (const float*)d_in[6];
  const float* Wq        = (const float*)d_in[7];
  const float* Wk        = (const float*)d_in[8];
  const float* Wv        = (const float*)d_in[9];
  const float* Wo        = (const float*)d_in[10];
  const float* Wpk       = (const float*)d_in[11];
  const float* F1        = (const float*)d_in[12];
  const float* fb1       = (const float*)d_in[13];
  const float* F2        = (const float*)d_in[14];
  const float* fb2       = (const float*)d_in[15];
  const unsigned char* maskp = (const unsigned char*)d_in[16];
  const int* opmach      = (const int*)d_in[17];

  _Float16* ws = (_Float16*)d_ws;
  size_t off = 0;
  _Float16* wih16 = ws + off; off += 768 * 256;
  _Float16* whh16 = ws + off; off += 768 * 256;
  _Float16* wq16  = ws + off; off += 256 * 256;
  _Float16* wk16  = ws + off; off += 256 * 256;
  _Float16* wv16  = ws + off; off += 256 * 256;
  _Float16* wo16  = ws + off; off += 256 * 256;
  _Float16* wpk16 = ws + off; off += 256 * 256;
  _Float16* f1p16 = ws + off; off += 256 * KPAD;
  _Float16* f216  = ws + off; off += 256 * 256;
  _Float16* kl16  = ws + off; off += (size_t)B_TOT * NNODE * 256;
  _Float16* ks16  = ws + off; off += (size_t)B_TOT * NNODE * 256;
  _Float16* vs16  = ws + off; off += (size_t)B_TOT * NNODE * 256;

  auto cv = [&](const float* s_, _Float16* d_, int n) {
    cvt_f32_f16<<<(n + 255) / 256, 256, 0, stream>>>(s_, d_, n);
  };
  cv(Wih, wih16, 768 * 256);
  cv(Whh, whh16, 768 * 256);
  cv(Wq, wq16, 256 * 256);
  cv(Wk, wk16, 256 * 256);
  cv(Wv, wv16, 256 * 256);
  cv(Wo, wo16, 256 * 256);
  cv(Wpk, wpk16, 256 * 256);
  cv(F2, f216, 256 * 256);
  cvt_pad_f16<<<(256 * KPAD + 255) / 256, 256, 0, stream>>>(F1, f1p16, 256, 262, KPAD);

  precompute_kernel<<<(B_TOT * NNODE) / BT, 256, 0, stream>>>(
      enc, wpk16, wk16, wv16, kl16, ks16, vs16);

  static_assert(sizeof(SmemDec) < 320 * 1024, "LDS budget exceeded");
  hipFuncSetAttribute(reinterpret_cast<const void*>(decoder_kernel),
                      hipFuncAttributeMaxDynamicSharedMemorySize, (int)sizeof(SmemDec));
  decoder_kernel<<<B_TOT / BT, 256, sizeof(SmemDec), stream>>>(
      enc, proc_t, start_tok, bih, bhh, fb1, fb2, maskp, opmach,
      wih16, whh16, wq16, wo16, f1p16, f216, kl16, ks16, vs16,
      (float*)d_out);
}
